// CacheFuser_73873437491748
// MI455X (gfx1250) — compile-verified, matrix operand order checked by reading, not run on hardware
//
#include <hip/hip_runtime.h>

typedef __bf16 bf16;
typedef __attribute__((ext_vector_type(16))) __bf16 v16bf;
typedef __attribute__((ext_vector_type(8)))  float  v8f;

#define L_      8
#define N_      4
#define MROWS   8192      // B*S rows per layer
#define H_      256
#define TILE_M  128
#define AP      264       // padded LDS row stride (halfs) for activation tiles
#define WP      40        // padded LDS col stride (halfs) for weight chunk (32 data + 8 pad)
#define KC      32        // K per WMMA chunk
#define NKC     (H_ / KC) // 8

struct __align__(16) Smem {
  bf16 X[TILE_M * AP];    // activation tile (A-matrix source), bf16 row-major
  bf16 Hs[TILE_M * AP];   // hidden tile between GEMM1 and GEMM2
  bf16 W[2][H_ * WP];     // double-buffered transposed weight K-chunk: W[buf][col*WP + k]
};

// ---- CDNA5 async global->LDS copy (ASYNCcnt) --------------------------------
__device__ inline void async_b128(void* lds_ptr, const void* gptr) {
  const unsigned lds = (unsigned)(uintptr_t)lds_ptr;            // LDS byte offset (addr[31:0])
  const unsigned long long ga = (unsigned long long)(uintptr_t)gptr;
  asm volatile("global_load_async_to_lds_b128 %0, %1, off" :: "v"(lds), "v"(ga) : "memory");
}
__device__ inline void wait_async0() {
  asm volatile("s_wait_asynccnt 0x0" ::: "memory");
}

// ---- fragment loads (ISA 7.12.2 16-bit layouts) ------------------------------
__device__ inline v16bf lds_a_frag(const bf16* base, int lane, int kbase) {
  const bf16* p = base + (lane & 15) * AP + kbase + (lane >> 4) * 8;
  v16bf a;
  ((uint4*)&a)[0] = *(const uint4*)(p);
  ((uint4*)&a)[1] = *(const uint4*)(p + 16);
  return a;
}
__device__ inline v16bf lds_b_frag(const bf16* wt, int lane) {
  const bf16* p = wt + (lane & 15) * WP + (lane >> 4) * 8;
  v16bf b;
  ((uint4*)&b)[0] = *(const uint4*)(p);
  ((uint4*)&b)[1] = *(const uint4*)(p + 16);
  return b;
}

// ---- staging -----------------------------------------------------------------
// async-copy one K-chunk of a pre-packed bf16 weight [256 cols][Ktot k] into LDS
__device__ inline void issue_w_chunk(bf16* Ws, const bf16* __restrict__ gwt,
                                     int Ktot, int kc, int tid) {
  #pragma unroll
  for (int it = 0; it < 2; ++it) {                 // 1024 x 16B segments / 512 thr
    const int seg = tid + it * 512;
    const int col = seg >> 2, s = seg & 3;
    async_b128(Ws + col * WP + s * 8,
               gwt + (size_t)col * Ktot + (size_t)kc * KC + s * 8);
  }
}
// async-copy a 128x256 bf16 activation tile into padded LDS rows
__device__ inline void issue_x_bf(bf16* Xs, const bf16* __restrict__ gx, int tid) {
  for (int seg = tid; seg < TILE_M * 32; seg += 512) {  // 16B segments
    const int r = seg >> 5, s = seg & 31;
    async_b128(Xs + r * AP + s * 8, gx + (size_t)r * H_ + s * 8);
  }
}
// f32 source needs conversion: VALU staging path
__device__ inline void stage_x_f32(bf16* Xs, const float* __restrict__ gx, int tid) {
  const int c4 = (tid & 63) * 4;
  for (int r = tid >> 6; r < TILE_M; r += 8) {
    const float4 v = *(const float4*)(gx + (size_t)r * H_ + c4);
    bf16* p = Xs + r * AP + c4;
    p[0] = (bf16)v.x; p[1] = (bf16)v.y; p[2] = (bf16)v.z; p[3] = (bf16)v.w;
  }
}

#define WMMA_BF16(A, B, C) \
  __builtin_amdgcn_wmma_f32_16x16x32_bf16(false, (A), false, (B), (short)0, (C), false, false)

// =============================================================================
// Weight pre-pack: f32 [L][Ktot][256] -> bf16 transposed [L][256][Ktot]
// grid (Ktot/32, 8, L), block (32,8)
// =============================================================================
__global__ __launch_bounds__(256) void pack_w(const float* __restrict__ src,
                                              bf16* __restrict__ dst, int Ktot) {
  __shared__ bf16 t[32][33];
  const int kb = blockIdx.x * 32, cb = blockIdx.y * 32, l = blockIdx.z;
  src += (size_t)l * Ktot * H_;
  dst += (size_t)l * Ktot * H_;
  const int x = threadIdx.x, y0 = threadIdx.y;
  #pragma unroll
  for (int y = y0; y < 32; y += 8)
    t[y][x] = (bf16)src[(size_t)(kb + y) * H_ + cb + x];      // coalesced read
  __syncthreads();
  #pragma unroll
  for (int y = y0; y < 32; y += 8)
    dst[(size_t)(cb + y) * Ktot + kb + x] = t[x][y];          // coalesced write
}

// =============================================================================
// Kernel A: align MLPs + edge-weighted aggregation over sharers.
//   Hsum = sum_n (ew[l,n]/N) * relu(X_n @ W1 + b1)   (GEMM2 hoisted: W2 shared over n)
//   agg  = Hsum @ W2 + (sum_n ew[l,n]/N) * b2        -> bf16 workspace
// grid: (MROWS/TILE_M, L, 2[k|v]); block: 512 (16 waves; wave = 16 rows x 128 cols)
// =============================================================================
__global__ __launch_bounds__(512) void cachefuser_align(
    const float* __restrict__ shrK, const float* __restrict__ shrV,
    const bf16* __restrict__ akw1t, const float* __restrict__ akb1,
    const bf16* __restrict__ akw2t, const float* __restrict__ akb2,
    const bf16* __restrict__ avw1t, const float* __restrict__ avb1,
    const bf16* __restrict__ avw2t, const float* __restrict__ avb2,
    const float* __restrict__ ew, bf16* __restrict__ agg)
{
  __shared__ Smem sm;
  const int tid = threadIdx.x;
  const int lane = tid & 31, wave = tid >> 5;
  const int row0 = (wave & 7) * 16;
  const int colBase = (wave >> 3) * 128;
  const int lo = lane & 15, hi = lane >> 4;
  const int tile = blockIdx.x, l = blockIdx.y, kv = blockIdx.z;
  const int rowG = tile * TILE_M;

  const float* shr = kv ? shrV : shrK;
  const bf16* w1t = (kv ? avw1t : akw1t) + (size_t)l * H_ * H_;
  const float* b1 = (kv ? avb1 : akb1) + (size_t)l * H_;
  const bf16* w2t = (kv ? avw2t : akw2t) + (size_t)l * H_ * H_;
  const float* b2 = (kv ? avb2 : akb2) + (size_t)l * H_;

  float b1r[8], b2r[8];
  #pragma unroll
  for (int ct = 0; ct < 8; ++ct) {
    const int col = colBase + ct * 16 + lo;
    b1r[ct] = b1[col];
    b2r[ct] = b2[col];
  }

  v8f hsum[8] = {};
  float sw = 0.f;

  for (int n = 0; n < N_; ++n) {
    __syncthreads();                                   // all waves done with sm.X / W bufs
    const float* gX = shr + (((size_t)l * N_ + n) * MROWS + rowG) * H_;
    stage_x_f32(sm.X, gX, tid);
    if (n + 1 < N_) {                                  // prefetch next sharer tile (HBM stream)
      const char* pn = (const char*)(shr + (((size_t)l * N_ + n + 1) * MROWS + rowG) * H_);
      __builtin_prefetch(pn + tid * 256, 0, 1);
    }
    v8f hacc[8] = {};
    issue_w_chunk(sm.W[0], w1t, H_, 0, tid);
    for (int kc = 0; kc < NKC; ++kc) {
      wait_async0();                                   // chunk kc landed (per wave)
      __syncthreads();                                 // visible to all; bufs free
      if (kc + 1 < NKC) issue_w_chunk(sm.W[(kc + 1) & 1], w1t, H_, kc + 1, tid);
      const v16bf a = lds_a_frag(sm.X + row0 * AP, lane, kc * KC);
      #pragma unroll
      for (int ct = 0; ct < 8; ++ct) {
        const v16bf b = lds_b_frag(sm.W[kc & 1] + (colBase + ct * 16) * WP, lane);
        hacc[ct] = WMMA_BF16(a, b, hacc[ct]);
      }
    }
    const float sn = ew[l * N_ + n] * (1.0f / N_);
    sw += sn;
    #pragma unroll
    for (int ct = 0; ct < 8; ++ct)
      #pragma unroll
      for (int i = 0; i < 8; ++i)
        hsum[ct][i] += sn * fmaxf(hacc[ct][i] + b1r[ct], 0.f);
  }

  // Hsum -> LDS (C layout: VGPR i, lane -> row row0+hi*8+i, col colBase+ct*16+lo)
  issue_w_chunk(sm.W[0], w2t, H_, 0, tid);
  #pragma unroll
  for (int ct = 0; ct < 8; ++ct) {
    const int col = colBase + ct * 16 + lo;
    #pragma unroll
    for (int i = 0; i < 8; ++i)
      sm.Hs[(row0 + hi * 8 + i) * AP + col] = (bf16)hsum[ct][i];
  }

  // GEMM2: agg = Hsum @ W2 + sw*b2
  v8f gacc[8] = {};
  for (int kc = 0; kc < NKC; ++kc) {
    wait_async0();
    __syncthreads();
    if (kc + 1 < NKC) issue_w_chunk(sm.W[(kc + 1) & 1], w2t, H_, kc + 1, tid);
    const v16bf a = lds_a_frag(sm.Hs + row0 * AP, lane, kc * KC);
    #pragma unroll
    for (int ct = 0; ct < 8; ++ct) {
      const v16bf b = lds_b_frag(sm.W[kc & 1] + (colBase + ct * 16) * WP, lane);
      gacc[ct] = WMMA_BF16(a, b, gacc[ct]);
    }
  }

  bf16* aggp = agg + (((size_t)kv * L_ + l) * MROWS + rowG) * H_;
  #pragma unroll
  for (int ct = 0; ct < 8; ++ct) {
    const int col = colBase + ct * 16 + lo;
    #pragma unroll
    for (int i = 0; i < 8; ++i)
      aggp[(size_t)(row0 + hi * 8 + i) * H_ + col] = (bf16)(gacc[ct][i] + sw * b2r[ct]);
  }
}

// =============================================================================
// Kernel B: gated fusion MLP.
//   h     = relu(recv @ W1[0:H] + agg @ W1[H:2H] + b1)   (K=512 in two phases)
//   delta = h @ W2 + b2
//   out   = recv + sigmoid(alpha/TAU) * delta            (f32 output)
// =============================================================================
__global__ __launch_bounds__(512) void cachefuser_fuse(
    const float* __restrict__ recvK, const float* __restrict__ recvV,
    const bf16* __restrict__ fkw1t, const float* __restrict__ fkb1,
    const bf16* __restrict__ fkw2t, const float* __restrict__ fkb2,
    const bf16* __restrict__ fvw1t, const float* __restrict__ fvb1,
    const bf16* __restrict__ fvw2t, const float* __restrict__ fvb2,
    const float* __restrict__ alpha, const bf16* __restrict__ agg,
    float* __restrict__ out)
{
  __shared__ Smem sm;
  const int tid = threadIdx.x;
  const int lane = tid & 31, wave = tid >> 5;
  const int row0 = (wave & 7) * 16;
  const int colBase = (wave >> 3) * 128;
  const int lo = lane & 15, hi = lane >> 4;
  const int tile = blockIdx.x, l = blockIdx.y, kv = blockIdx.z;
  const int rowG = tile * TILE_M;

  const float* recv = kv ? recvV : recvK;
  const bf16* w1t = (kv ? fvw1t : fkw1t) + (size_t)l * (2 * H_) * H_;  // [256][512]
  const float* b1 = (kv ? fvb1 : fkb1) + (size_t)l * H_;
  const bf16* w2t = (kv ? fvw2t : fkw2t) + (size_t)l * H_ * H_;
  const float* b2 = (kv ? fvb2 : fkb2) + (size_t)l * H_;

  const float* gR = recv + ((size_t)l * MROWS + rowG) * H_;
  const bf16*  gA = agg + (((size_t)kv * L_ + l) * MROWS + rowG) * H_;
  const float  av = alpha[l];
  const float  gate = 1.0f / (1.0f + __expf(-av * 2.0f));  // sigmoid(alpha/0.5)

  float b1r[8], b2r[8];
  #pragma unroll
  for (int ct = 0; ct < 8; ++ct) {
    const int col = colBase + ct * 16 + lo;
    b1r[ct] = b1[col];
    b2r[ct] = b2[col];
  }

  // GEMM1 over K=512: phase 0 = receiver vs W1 k[0,256); phase 1 = agg vs k[256,512)
  v8f hacc[8] = {};
  for (int ph = 0; ph < 2; ++ph) {
    __syncthreads();                                   // all waves done reading sm.X
    if (ph == 0) stage_x_f32(sm.X, gR, tid);           // f32 -> bf16 via VALU
    else         issue_x_bf (sm.X, gA, tid);           // bf16, L2-resident -> async copy
    issue_w_chunk(sm.W[0], w1t, 2 * H_, ph * NKC, tid);
    for (int kc = 0; kc < NKC; ++kc) {
      wait_async0();
      __syncthreads();
      if (kc + 1 < NKC) issue_w_chunk(sm.W[(kc + 1) & 1], w1t, 2 * H_, ph * NKC + kc + 1, tid);
      const v16bf a = lds_a_frag(sm.X + row0 * AP, lane, kc * KC);
      #pragma unroll
      for (int ct = 0; ct < 8; ++ct) {
        const v16bf b = lds_b_frag(sm.W[kc & 1] + (colBase + ct * 16) * WP, lane);
        hacc[ct] = WMMA_BF16(a, b, hacc[ct]);
      }
    }
  }

  // bias + relu -> Hs
  issue_w_chunk(sm.W[0], w2t, H_, 0, tid);
  #pragma unroll
  for (int ct = 0; ct < 8; ++ct) {
    const int col = colBase + ct * 16 + lo;
    #pragma unroll
    for (int i = 0; i < 8; ++i)
      sm.Hs[(row0 + hi * 8 + i) * AP + col] = (bf16)fmaxf(hacc[ct][i] + b1r[ct], 0.f);
  }

  // GEMM2: delta = h @ W2
  v8f dacc[8] = {};
  for (int kc = 0; kc < NKC; ++kc) {
    wait_async0();
    __syncthreads();
    if (kc + 1 < NKC) issue_w_chunk(sm.W[(kc + 1) & 1], w2t, H_, kc + 1, tid);
    const v16bf a = lds_a_frag(sm.Hs + row0 * AP, lane, kc * KC);
    #pragma unroll
    for (int ct = 0; ct < 8; ++ct) {
      const v16bf b = lds_b_frag(sm.W[kc & 1] + (colBase + ct * 16) * WP, lane);
      dacc[ct] = WMMA_BF16(a, b, dacc[ct]);
    }
  }

  // epilogue: out = recv + gate * (delta + b2)
  float* op = out + (((size_t)kv * L_ + l) * MROWS + rowG) * H_;
  #pragma unroll
  for (int ct = 0; ct < 8; ++ct) {
    const int col = colBase + ct * 16 + lo;
    #pragma unroll
    for (int i = 0; i < 8; ++i) {
      const size_t idx = (size_t)(row0 + hi * 8 + i) * H_ + col;
      op[idx] = gR[idx] + gate * (dacc[ct][i] + b2r[ct]);
    }
  }
}

extern "C" void kernel_launch(void* const* d_in, const int* in_sizes, int n_in,
                              void* d_out, int out_size, void* d_ws, size_t ws_size,
                              hipStream_t stream) {
  (void)in_sizes; (void)n_in; (void)out_size; (void)ws_size;
  const float* receiver_k = (const float*)d_in[0];
  const float* receiver_v = (const float*)d_in[1];
  const float* sharer_k   = (const float*)d_in[2];
  const float* sharer_v   = (const float*)d_in[3];
  const float* edge_w     = (const float*)d_in[4];
  const float* alpha      = (const float*)d_in[5];
  const float* ak_w1 = (const float*)d_in[6];
  const float* ak_b1 = (const float*)d_in[7];
  const float* ak_w2 = (const float*)d_in[8];
  const float* ak_b2 = (const float*)d_in[9];
  const float* av_w1 = (const float*)d_in[10];
  const float* av_b1 = (const float*)d_in[11];
  const float* av_w2 = (const float*)d_in[12];
  const float* av_b2 = (const float*)d_in[13];
  const float* fk_w1 = (const float*)d_in[14];
  const float* fk_b1 = (const float*)d_in[15];
  const float* fk_w2 = (const float*)d_in[16];
  const float* fk_b2 = (const float*)d_in[17];
  const float* fv_w1 = (const float*)d_in[18];
  const float* fv_b1 = (const float*)d_in[19];
  const float* fv_w2 = (const float*)d_in[20];
  const float* fv_b2 = (const float*)d_in[21];

  // workspace layout (bf16): [agg 2*L*M*H | packed transposed weights]
  bf16* agg = (bf16*)d_ws;
  const size_t aggElems = (size_t)2 * L_ * MROWS * H_;     // 64 MB
  const size_t m256 = (size_t)L_ * H_ * H_;                // 256x256 per layer
  const size_t m512 = (size_t)L_ * 2 * H_ * H_;            // 512x256 per layer
  bf16* akw1t = agg + aggElems;
  bf16* avw1t = akw1t + m256;
  bf16* akw2t = avw1t + m256;
  bf16* avw2t = akw2t + m256;
  bf16* fkw2t = avw2t + m256;
  bf16* fvw2t = fkw2t + m256;
  bf16* fkw1t = fvw2t + m256;
  bf16* fvw1t = fkw1t + m512;

  dim3 pb(32, 8);
  pack_w<<<dim3(H_ / 32, H_ / 32, L_), pb, 0, stream>>>(ak_w1, akw1t, H_);
  pack_w<<<dim3(H_ / 32, H_ / 32, L_), pb, 0, stream>>>(av_w1, avw1t, H_);
  pack_w<<<dim3(H_ / 32, H_ / 32, L_), pb, 0, stream>>>(ak_w2, akw2t, H_);
  pack_w<<<dim3(H_ / 32, H_ / 32, L_), pb, 0, stream>>>(av_w2, avw2t, H_);
  pack_w<<<dim3(H_ / 32, H_ / 32, L_), pb, 0, stream>>>(fk_w2, fkw2t, H_);
  pack_w<<<dim3(H_ / 32, H_ / 32, L_), pb, 0, stream>>>(fv_w2, fvw2t, H_);
  pack_w<<<dim3(2 * H_ / 32, H_ / 32, L_), pb, 0, stream>>>(fk_w1, fkw1t, 2 * H_);
  pack_w<<<dim3(2 * H_ / 32, H_ / 32, L_), pb, 0, stream>>>(fv_w1, fvw1t, 2 * H_);

  dim3 grid(MROWS / TILE_M, L_, 2);
  dim3 block(512);

  cachefuser_align<<<grid, block, 0, stream>>>(
      sharer_k, sharer_v,
      akw1t, ak_b1, akw2t, ak_b2,
      avw1t, av_b1, avw2t, av_b2,
      edge_w, agg);

  cachefuser_fuse<<<grid, block, 0, stream>>>(
      receiver_k, receiver_v,
      fkw1t, fk_b1, fkw2t, fk_b2,
      fvw1t, fv_b1, fvw2t, fv_b2,
      alpha, agg, (float*)d_out);
}
